// SimpleTumorGNN_2611340116443
// MI455X (gfx1250) — matrix operand order repaired and checked by reading, not compiled, not run on hardware
//
#include <hip/hip_runtime.h>
#include <hip/hip_fp16.h>

// CDNA5 / gfx1250. wave32. WMMA f32_16x16x32_f16 for the dense GEMMs with
// register-resident pre-packed B fragments; global f32 atomics for all
// segment reductions.

typedef __attribute__((ext_vector_type(16))) _Float16 v16h;
typedef __attribute__((ext_vector_type(8)))  _Float16 v8h;
typedef __attribute__((ext_vector_type(8)))  float    v8f;

#define GS(i, n) for (long i = (long)blockIdx.x * blockDim.x + threadIdx.x; i < (n); \
                      i += (long)gridDim.x * blockDim.x)

__device__ __forceinline__ void atomicMaxF32(float* a, float v) {
  // classic sign-split trick; location initialized to -inf
  if (v >= 0.0f) atomicMax((int*)a, __float_as_int(v));
  else           atomicMin((unsigned int*)a, __float_as_uint(v));
}

// ---------------------------------------------------------------- utility
__global__ void k_fill(float* __restrict__ p, float v, long n) { GS(i, n) p[i] = v; }

// ---------------------------------------------------------------- GAT front
// xw = x (N,5) @ gat_w (5,128)
__global__ void k_xw(const float* __restrict__ x, const float* __restrict__ w,
                     float* __restrict__ xw, long n) {
  GS(i, n * 128) {
    int j = (int)(i & 127); long r = i >> 7;
    const float* xr = x + r * 5;
    float s = 0.f;
#pragma unroll
    for (int t = 0; t < 5; ++t) s += xr[t] * w[t * 128 + j];
    xw[i] = s;
  }
}

// a_src/a_dst per (node, head)
__global__ void k_att(const float* __restrict__ xw, const float* __restrict__ as,
                      const float* __restrict__ ad, float* __restrict__ a_src,
                      float* __restrict__ a_dst, long n) {
  GS(i, n * 2) {
    int h = (int)(i & 1); long r = i >> 1;
    const float* p = xw + r * 128 + h * 64;
    float s1 = 0.f, s2 = 0.f;
    for (int c = 0; c < 64; ++c) { float v = p[c]; s1 += v * as[h * 64 + c]; s2 += v * ad[h * 64 + c]; }
    a_src[i] = s1; a_dst[i] = s2;
  }
}

// e = leaky_relu(a_src[src]+a_dst[dst]); segment max into emax[dst]
__global__ void k_edge_e(const int* __restrict__ src, const int* __restrict__ dst,
                         const float* __restrict__ a_src, const float* __restrict__ a_dst,
                         float* __restrict__ e, float* __restrict__ emax, long E) {
  GS(i, E * 2) {
    int h = (int)(i & 1); long eid = i >> 1;
    int s = src[eid], d = dst[eid];
    float v = a_src[(long)s * 2 + h] + a_dst[(long)d * 2 + h];
    v = v > 0.f ? v : 0.2f * v;
    e[i] = v;
    atomicMaxF32(&emax[(long)d * 2 + h], v);
  }
}

// ex = exp(e - emax[dst]); denom[dst] += ex   (stored in place of e)
__global__ void k_edge_exp(const int* __restrict__ dst, float* __restrict__ e,
                           const float* __restrict__ emax, float* __restrict__ denom, long E) {
  GS(i, E * 2) {
    int h = (int)(i & 1); long eid = i >> 1; int d = dst[eid];
    float v = expf(e[i] - emax[(long)d * 2 + h]);
    e[i] = v;
    atomicAdd(&denom[(long)d * 2 + h], v);
  }
}

__global__ void k_deg(const int* __restrict__ dst, float* __restrict__ deg, long E) {
  GS(i, E) atomicAdd(&deg[dst[i]], 1.f);
}

// h1[dst] += xw[src] * alpha,  alpha = ex / (denom[dst]+1e-16)
__global__ void k_gat_scatter(const int* __restrict__ src, const int* __restrict__ dst,
                              const float* __restrict__ xw, const float* __restrict__ ex,
                              const float* __restrict__ denom, float* __restrict__ h1, long E) {
  GS(i, E * 128) {
    int ch = (int)(i & 127); long eid = i >> 7; int h = ch >> 6;
    int s = src[eid], d = dst[eid];
    float alpha = ex[eid * 2 + h] / (denom[(long)d * 2 + h] + 1e-16f);
    atomicAdd(&h1[(long)d * 128 + ch], xw[(long)s * 128 + ch] * alpha);
  }
}

// ---------------------------------------------------------------- BatchNorm
// per-channel sum and sum-of-squares; blockDim.x == C, 512 rows per block
__global__ void k_bn_stats(const float* __restrict__ h, long n, float* __restrict__ sums) {
  int C = blockDim.x; int c = threadIdx.x;
  long r0 = (long)blockIdx.x * 512;
  long r1 = r0 + 512; if (r1 > n) r1 = n;
  float s = 0.f, s2 = 0.f;
  for (long r = r0; r < r1; ++r) { float v = h[r * C + c]; s += v; s2 += v * v; }
  atomicAdd(&sums[c], s);
  atomicAdd(&sums[C + c], s2);
}

// BN + ELU in place; optional f16 copy into out16 at column offset col16
__global__ void k_bn_apply(float* __restrict__ h, long n, int cshift,
                           const float* __restrict__ g, const float* __restrict__ b,
                           const float* __restrict__ sums,
                           _Float16* __restrict__ out16, int ld16, int col16) {
  const int C = 1 << cshift; const int mask = C - 1;
  const float invN = 1.0f / (float)n;
  GS(i, n << cshift) {
    int c = (int)(i & mask); long r = i >> cshift;
    float m = sums[c] * invN;
    float var = sums[C + c] * invN - m * m;
    float y = (h[i] - m) * rsqrtf(var + 1e-5f) * g[c] + b[c];
    y = y > 0.f ? y : (expf(y) - 1.f);   // ELU
    h[i] = y;
    if (out16) out16[r * (long)ld16 + col16 + c] = (_Float16)y;
  }
}

// ---------------------------------------------------------------- SAGE
__global__ void k_sage_scatter(const int* __restrict__ src, const int* __restrict__ dst,
                               const float* __restrict__ h1, float* __restrict__ agg, long E) {
  GS(i, E * 128) {
    int ch = (int)(i & 127); long eid = i >> 7;
    int s = src[eid], d = dst[eid];
    atomicAdd(&agg[(long)d * 128 + ch], h1[(long)s * 128 + ch]);
  }
}

// agg/deg -> f16 into hcat columns [0,128)
__global__ void k_sage_div(const float* __restrict__ agg, const float* __restrict__ deg,
                           _Float16* __restrict__ hcat, long n) {
  GS(i, n * 128) {
    int ch = (int)(i & 127); long r = i >> 7;
    float v = agg[i] / fmaxf(deg[r], 1.f);
    hcat[r * 256 + ch] = (_Float16)v;
  }
}

// weights -> f16 row-major: wcat = [sage_wl ; sage_wr] (256x64), gw16 = gcn_w (64x64)
__global__ void k_wcat(const float* __restrict__ wl, const float* __restrict__ wr,
                       const float* __restrict__ gw, _Float16* __restrict__ wcat,
                       _Float16* __restrict__ gw16) {
  GS(i, 256 * 64 + 64 * 64) {
    if (i < 256 * 64) {
      long r = i >> 6, c = i & 63;
      float v = r < 128 ? wl[r * 64 + c] : wr[(r - 128) * 64 + c];
      wcat[i] = (_Float16)v;
    } else {
      long j = i - 256 * 64;
      gw16[j] = (_Float16)gw[j];
    }
  }
}

// Pack row-major B (K x 64, f16) into WMMA lane-fragment order so each lane's
// 16-element B fragment is one contiguous 32-byte chunk:
//   Bp[kt][nt][lane][j] = B[kt*32 + (j>>3)*16 + (lane>>4)*8 + (j&7)][nt*16 + (lane&15)]
__global__ void k_bpack(const _Float16* __restrict__ B, _Float16* __restrict__ Bp, int KT) {
  GS(i, (long)KT * 2048) {
    int j  = (int)(i & 15);
    int L  = (int)((i >> 4) & 31);
    int nt = (int)((i >> 9) & 3);
    int kt = (int)(i >> 11);
    int col = nt * 16 + (L & 15);
    int k   = kt * 32 + (j >> 3) * 16 + (L >> 4) * 8 + (j & 7);
    Bp[i] = B[(long)k * 64 + col];
  }
}

// ---------------------------------------------------------------- WMMA GEMM
// C(M,64) = A(M,K) @ B(K,64), f16 in, f32 out, A row-major, B pre-packed.
// KT = K/32 tiles (templated so B fragments live in registers).
// MT m-tiles per block; grid.x = M/(16*MT); block = 128 (4 waves, one n-tile each).
// EXEC is all-ones by construction (exact grid, no divergence).
template <int KT, int MT>
__global__ void k_gemm_wmma(const _Float16* __restrict__ A, const _Float16* __restrict__ Bp,
                            float* __restrict__ C) {
  const int lane = threadIdx.x & 31;
  const int wv   = threadIdx.x >> 5;     // n-tile 0..3
  const int r16  = lane & 15;
  const int hi   = lane >> 4;
  const int K    = KT * 32;

  // preload all B fragments for this wave's n-tile: one 32B vector load each
  v16h bf[KT];
#pragma unroll
  for (int kt = 0; kt < KT; ++kt)
    bf[kt] = *(const v16h*)(Bp + (((long)kt * 4 + wv) * 32 + lane) * 16);

  const long m0 = (long)blockIdx.x * (16 * MT);
#pragma unroll
  for (int mt = 0; mt < MT; ++mt) {
    const long arow = m0 + mt * 16 + r16;
    const _Float16* Ab = A + arow * K + hi * 8;
    v8f acc = {};
#pragma unroll
    for (int kt = 0; kt < KT; ++kt) {
      // A fragment: lanes<16 hold K {0..7,16..23}+kt*32; lanes>=16 shifted by 8
      v8h alo = *(const v8h*)(Ab + kt * 32);
      v8h ahi = *(const v8h*)(Ab + kt * 32 + 16);
      v16h a;
#pragma unroll
      for (int j = 0; j < 8; ++j) { a[j] = alo[j]; a[8 + j] = ahi[j]; }
      acc = __builtin_amdgcn_wmma_f32_16x16x32_f16(false, a, false, bf[kt],
                                                   (short)0, acc, false, false);
    }
    // C/D layout: VGPR r -> row r + hi*8, col = wv*16 + r16
    float* Cp = C + (m0 + mt * 16 + hi * 8) * 64 + wv * 16 + r16;
#pragma unroll
    for (int r = 0; r < 8; ++r) Cp[(long)r * 64] = acc[r];
  }
}

// ---------------------------------------------------------------- GCN
__global__ void k_gcn_self(const float* __restrict__ hw, const float* __restrict__ deg,
                           float* __restrict__ h3, long n) {
  GS(i, n * 64) {
    long r = i >> 6;
    float di2 = 1.f / (deg[r] + 1.f);        // dinv*dinv with self loop
    h3[i] = hw[i] * di2;
  }
}

__global__ void k_gcn_scatter(const int* __restrict__ src, const int* __restrict__ dst,
                              const float* __restrict__ hw, const float* __restrict__ deg,
                              float* __restrict__ h3, long E) {
  GS(i, E * 64) {
    int c = (int)(i & 63); long eid = i >> 6;
    int s = src[eid], d = dst[eid];
    float coef = rsqrtf(deg[s] + 1.f) * rsqrtf(deg[d] + 1.f);
    atomicAdd(&h3[(long)d * 64 + c], hw[(long)s * 64 + c] * coef);
  }
}

// ---------------------------------------------------------------- pooling + head
__global__ void k_pool(const float* __restrict__ h, const int* __restrict__ batch,
                       float* __restrict__ psum, float* __restrict__ pmax,
                       float* __restrict__ cnt, long n) {
  GS(i, n * 64) {
    int c = (int)(i & 63); long r = i >> 6; int g = batch[r];
    float v = h[i];
    atomicAdd(&psum[(long)g * 64 + c], v);
    atomicMaxF32(&pmax[(long)g * 64 + c], v);
    if (c == 0) atomicAdd(&cnt[g], 1.f);
  }
}

__global__ void k_fc1(const float* __restrict__ psum, const float* __restrict__ pmax,
                      const float* __restrict__ cnt, const float* __restrict__ w1,
                      const float* __restrict__ b1, float* __restrict__ zt, int G) {
  GS(i, (long)G * 64) {
    int j = (int)(i & 63); int g = (int)(i >> 6);
    float invc = 1.f / fmaxf(cnt[g], 1.f);
    float s = b1[j];
    for (int c = 0; c < 64; ++c) s += psum[g * 64 + c] * invc * w1[c * 64 + j];
    for (int c = 0; c < 64; ++c) {
      float v = pmax[g * 64 + c];
      if (!(v > -3.0e38f)) v = 0.f;          // where(isfinite, ., 0) for -inf init
      s += v * w1[(64 + c) * 64 + j];
    }
    zt[i] = s > 0.f ? s : 0.f;               // ReLU
  }
}

__global__ void k_fc2(const float* __restrict__ zt, const float* __restrict__ w2,
                      const float* __restrict__ b2, float* __restrict__ out, int G) {
  GS(i, (long)G * 4) {
    int k = (int)(i & 3); int g = (int)(i >> 2);
    float s = b2[k];
    for (int j = 0; j < 64; ++j) s += zt[g * 64 + j] * w2[j * 4 + k];
    out[i] = s;
  }
}

// ---------------------------------------------------------------- launch
extern "C" void kernel_launch(void* const* d_in, const int* in_sizes, int n_in,
                              void* d_out, int out_size, void* d_ws, size_t ws_size,
                              hipStream_t stream) {
  const float* x       = (const float*)d_in[0];
  const int*   ei      = (const int*)  d_in[1];
  const int*   batch   = (const int*)  d_in[2];
  const float* gat_w   = (const float*)d_in[3];
  const float* att_src = (const float*)d_in[4];
  const float* att_dst = (const float*)d_in[5];
  // d_in[6] gat_b, d_in[10] sage_bl, d_in[15] gcn_b cancel inside BatchNorm (and are zero)
  const float* bn1_g = (const float*)d_in[7];  const float* bn1_b = (const float*)d_in[8];
  const float* sage_wl = (const float*)d_in[9];
  const float* sage_wr = (const float*)d_in[11];
  const float* bn2_g = (const float*)d_in[12]; const float* bn2_b = (const float*)d_in[13];
  const float* gcn_w = (const float*)d_in[14];
  const float* bn3_g = (const float*)d_in[16]; const float* bn3_b = (const float*)d_in[17];
  const float* w1 = (const float*)d_in[18]; const float* b1 = (const float*)d_in[19];
  const float* w2 = (const float*)d_in[20]; const float* b2 = (const float*)d_in[21];
  float* out = (float*)d_out;

  const long N = in_sizes[0] / 5;        // 100000 (multiple of 80)
  const long E = in_sizes[1] / 2;        // 1600000
  const int  G = out_size / 4;           // 64
  const int* srcI = ei;
  const int* dstI = ei + E;

  // -------- workspace layout (lifetime-aliased) --------
  float* xw   = (float*)d_ws;                 // N*128   (reused: agg, then hw)
  float* h1   = xw + (size_t)N * 128;         // N*128   (reused: h3)
  float* ebuf = h1 + (size_t)N * 128;         // E*2     (e -> ex in place)
  float* h2   = ebuf + (size_t)E * 2;         // N*64
  _Float16* hcat16 = (_Float16*)(h2 + (size_t)N * 64);   // N*256 halves (reused: h2_16)
  // f16 weight area first (keeps 32B alignment for packed fragments)
  _Float16* wcatRM  = hcat16 + (size_t)N * 256;   // 256*64 row-major
  _Float16* gwRM    = wcatRM + 256 * 64;          // 64*64 row-major
  _Float16* wcatPK  = gwRM + 64 * 64;             // 256*64 packed fragments
  _Float16* gwPK    = wcatPK + 256 * 64;          // 64*64 packed fragments
  float* smallf = (float*)(gwPK + 64 * 64);
  float* a_srcv = smallf;              // N*2
  float* a_dstv = a_srcv + N * 2;      // N*2
  float* emax   = a_dstv + N * 2;      // N*2
  float* denom  = emax   + N * 2;      // N*2
  float* deg    = denom  + N * 2;      // N
  float* bn1s   = deg + N;             // 256
  float* bn2s   = bn1s + 256;          // 128
  float* bn3s   = bn2s + 128;          // 128
  float* psum   = bn3s + 128;          // G*64
  float* pmax   = psum + (size_t)G * 64;   // G*64
  float* cnt    = pmax + (size_t)G * 64;   // G
  float* zt     = cnt + G;                 // G*64
  float* agg = xw;   // alias after GAT scatter
  float* hw  = xw;   // alias after SAGE
  float* h3  = h1;   // alias after hcat16 is built

  const float NEG_INF = -__builtin_huge_valf();
  auto g1 = [](long n) -> dim3 {
    long bl = (n + 255) / 256; if (bl > 262144) bl = 262144; return dim3((unsigned)bl);
  };

  // -------- init accumulators (ws is poisoned; must re-init every call) --------
  k_fill<<<g1(N * 2), 256, 0, stream>>>(emax, NEG_INF, N * 2);
  k_fill<<<g1(N * 2), 256, 0, stream>>>(denom, 0.f, N * 2);
  k_fill<<<g1(N), 256, 0, stream>>>(deg, 0.f, N);
  k_fill<<<g1(N * 128), 256, 0, stream>>>(h1, 0.f, N * 128);
  k_fill<<<g1(512), 256, 0, stream>>>(bn1s, 0.f, 512);          // bn1s+bn2s+bn3s contiguous
  k_fill<<<g1((long)G * 64), 256, 0, stream>>>(psum, 0.f, (long)G * 64);
  k_fill<<<g1((long)G * 64), 256, 0, stream>>>(pmax, NEG_INF, (long)G * 64);
  k_fill<<<g1(G), 256, 0, stream>>>(cnt, 0.f, G);

  // -------- weights to f16 + pack into WMMA fragment order --------
  k_wcat<<<g1(256 * 64 + 64 * 64), 256, 0, stream>>>(sage_wl, sage_wr, gcn_w, wcatRM, gwRM);
  k_bpack<<<g1(8L * 2048), 256, 0, stream>>>(wcatRM, wcatPK, 8);
  k_bpack<<<g1(2L * 2048), 256, 0, stream>>>(gwRM, gwPK, 2);

  // -------- GATConv --------
  k_xw<<<g1(N * 128), 256, 0, stream>>>(x, gat_w, xw, N);
  k_att<<<g1(N * 2), 256, 0, stream>>>(xw, att_src, att_dst, a_srcv, a_dstv, N);
  k_edge_e<<<g1(E * 2), 256, 0, stream>>>(srcI, dstI, a_srcv, a_dstv, ebuf, emax, E);
  k_edge_exp<<<g1(E * 2), 256, 0, stream>>>(dstI, ebuf, emax, denom, E);
  k_deg<<<g1(E), 256, 0, stream>>>(dstI, deg, E);
  k_gat_scatter<<<g1(E * 128), 256, 0, stream>>>(srcI, dstI, xw, ebuf, denom, h1, E);
  k_bn_stats<<<dim3((unsigned)((N + 511) / 512)), 128, 0, stream>>>(h1, N, bn1s);
  k_bn_apply<<<g1(N * 128), 256, 0, stream>>>(h1, N, 7, bn1_g, bn1_b, bn1s,
                                              hcat16, 256, 128);   // h1 -> hcat[:,128:256]

  // -------- SAGEConv: h2 = [agg | h1] @ [wl ; wr]  (WMMA, K=256) --------
  k_fill<<<g1(N * 128), 256, 0, stream>>>(agg, 0.f, N * 128);     // xw dead, reuse
  k_sage_scatter<<<g1(E * 128), 256, 0, stream>>>(srcI, dstI, h1, agg, E);
  k_sage_div<<<g1(N * 128), 256, 0, stream>>>(agg, deg, hcat16, N);  // -> hcat[:,0:128]
  k_gemm_wmma<8, 5><<<dim3((unsigned)(N / 80)), 128, 0, stream>>>(hcat16, wcatPK, h2);
  k_bn_stats<<<dim3((unsigned)((N + 511) / 512)), 64, 0, stream>>>(h2, N, bn2s);
  k_bn_apply<<<g1(N * 64), 256, 0, stream>>>(h2, N, 6, bn2_g, bn2_b, bn2s,
                                             hcat16, 64, 0);      // h2_16 reuses hcat16

  // -------- GCNConv: hw = h2 @ gcn_w (WMMA, K=64), then sym-norm scatter --------
  k_gemm_wmma<2, 5><<<dim3((unsigned)(N / 80)), 128, 0, stream>>>(hcat16, gwPK, hw);
  k_gcn_self<<<g1(N * 64), 256, 0, stream>>>(hw, deg, h3, N);
  k_gcn_scatter<<<g1(E * 64), 256, 0, stream>>>(srcI, dstI, hw, deg, h3, E);
  k_bn_stats<<<dim3((unsigned)((N + 511) / 512)), 64, 0, stream>>>(h3, N, bn3s);
  k_bn_apply<<<g1(N * 64), 256, 0, stream>>>(h3, N, 6, bn3_g, bn3_b, bn3s,
                                             (_Float16*)nullptr, 0, 0);

  // -------- pooling + classifier --------
  k_pool<<<g1(N * 64), 256, 0, stream>>>(h3, batch, psum, pmax, cnt, N);
  k_fc1<<<g1((long)G * 64), 256, 0, stream>>>(psum, pmax, cnt, w1, b1, zt, G);
  k_fc2<<<g1((long)G * 4), 256, 0, stream>>>(zt, w2, b2, out, G);
}
